// SpinorGate_52192442581442
// MI455X (gfx1250) — matrix operand order repaired and checked by reading, not compiled, not run on hardware
//
#include <hip/hip_runtime.h>

typedef __attribute__((ext_vector_type(16))) __bf16 v16bf;
typedef __attribute__((ext_vector_type(8)))  float  v8f;
typedef int v4i_vs __attribute__((vector_size(4 * sizeof(int))));

#define K_LEVELS 8
#define NBLK     128      // N = D / BS
#define BSZ      8
#define DTOT     1024
#define MTOT     16384    // B * S
#define KTOT     2048     // 2 * D

#if __has_builtin(__builtin_amdgcn_global_load_async_to_lds_b128) && __has_builtin(__builtin_amdgcn_s_wait_asynccnt)
#define USE_ASYNC 1
#else
#define USE_ASYNC 0
#endif

__device__ __forceinline__ void g2l_16B(const void* src, void* lds_dst) {
#if USE_ASYNC
  __builtin_amdgcn_global_load_async_to_lds_b128(
      (__attribute__((address_space(1))) v4i_vs*)src,
      (__attribute__((address_space(3))) v4i_vs*)lds_dst, 0, 0);
#else
  *(float4*)lds_dst = *(const float4*)src;
#endif
}

#if USE_ASYNC
#define WAIT_ASYNC(n) __builtin_amdgcn_s_wait_asynccnt(n)
#else
#define WAIT_ASYNC(n)
#endif

// ---------------------------------------------------------------------------
// Kernel 0: convert W_gate (2048x1024 f32, row-major) into bf16 tiles laid out
// in WMMA B-fragment order. Tile = (32 K) x (16 cols); tile id
// t = kchunk*64 + n16; within a tile, lane l owns 16 contiguous bf16:
//   l < 16 : col = n16*16 + l,      K = kchunk*32 + 0..15
//   l >= 16: col = n16*16 + l - 16, K = kchunk*32 + 16..31
// so a B fragment is 2 contiguous ds_load_b128 and the per-WG chunk slice
// (8 tiles = 8KB) is contiguous for async b128 copies.
// ---------------------------------------------------------------------------
__global__ __launch_bounds__(256) void wconv_kernel(const float* __restrict__ W,
                                                    __bf16* __restrict__ Wb) {
  int e = blockIdx.x * 256 + threadIdx.x;  // 0 .. KTOT*DTOT-1
  int t    = e >> 9;                       // tile (512 halves each)
  int r    = e & 511;
  int lane = r >> 4;
  int h    = r & 15;
  int kchunk = t >> 6;
  int n16    = t & 63;
  int col = n16 * 16 + (lane & 15);
  int k   = kchunk * 32 + (lane >> 4) * 16 + h;
  Wb[e] = (__bf16)W[(size_t)k * DTOT + col];
}

// ---------------------------------------------------------------------------
// Kernel 1: Cayley transform. One block per (level k, block n); 16 working
// lanes do Gaussian elimination (no pivoting; diagonally dominant) on the
// 16x32 augmented real-block system
//   [[I+gAr, -gAi],[gAi, I+gAr]] X = [[I-gAr, gAi],[-gAi, I-gAr]]
// U_r = X[:8,:8], U_i = X[8:,:8].
// ---------------------------------------------------------------------------
__global__ void cayley_kernel(const float* __restrict__ A_real,
                              const float* __restrict__ A_imag,
                              float* __restrict__ Ur, float* __restrict__ Ui) {
  __shared__ float Aug[16][33];
  const int idx = blockIdx.x;          // k * NBLK + n
  const int k   = idx >> 7;
  const int n   = idx & (NBLK - 1);
  const float g = (float)k * (1.0f / 7.0f);
  const int t   = threadIdx.x;
  const float* Ar = A_real + n * 64;
  const float* Ai = A_imag + n * 64;

  if (t < 16) {
    const int  r     = t & 7;
    const bool lower = (t >= 8);
    for (int c = 0; c < 8; ++c) {
      float ar  = 0.5f * g * (Ar[r * 8 + c] - Ar[c * 8 + r]);  // skew part
      float ai  = 0.5f * g * (Ai[r * 8 + c] + Ai[c * 8 + r]);  // sym part
      float eye = (r == c) ? 1.0f : 0.0f;
      if (!lower) {
        Aug[t][c]      = eye + ar;   Aug[t][8 + c]  = -ai;
        Aug[t][16 + c] = eye - ar;   Aug[t][24 + c] = ai;
      } else {
        Aug[t][c]      = ai;         Aug[t][8 + c]  = eye + ar;
        Aug[t][16 + c] = -ai;        Aug[t][24 + c] = eye - ar;
      }
    }
  }

  for (int p = 0; p < 16; ++p) {
    __syncthreads();
    if (t == p) {
      float inv = 1.0f / Aug[p][p];
      for (int j = p; j < 32; ++j) Aug[p][j] *= inv;
    }
    __syncthreads();
    if (t < 16 && t != p) {
      float f = Aug[t][p];
      for (int j = p; j < 32; ++j) Aug[t][j] -= f * Aug[p][j];
    }
  }
  __syncthreads();

  if (t < 8) {
    for (int i = 0; i < 8; ++i) {
      Ur[idx * 64 + t * 8 + i] = Aug[t][16 + i];
      Ui[idx * 64 + t * 8 + i] = Aug[8 + t][16 + i];
    }
  }
}

// ---------------------------------------------------------------------------
// Kernel 2: gate GEMM (bf16 WMMA, fp32 accumulate) fused with sigmoid,
// 8-column mean-pool and level quantization.
// C[16384 x 1024] = Xcat[16384 x 2048] @ W[2048 x 1024]
// WG = 256 threads = 8 waves, tile 128(M) x 128(N); waves 4(M) x 2(N);
// each wave: 32x64 = 2x4 16x16 tiles -> 8 WMMA per K-chunk per wave.
// K-chunks of 32, double-buffered LDS via async LDS copies:
//   A: fp32 128x32 tile (pitch 36 floats), converted to bf16 at pack time
//   B: pre-converted bf16 fragment-order tiles -> 2 ds_load_b128 per fragment
// ---------------------------------------------------------------------------
#define TM 128
#define TN 128
#define TK 32
#define AS_STRIDE 36     // 144B row pitch: 16B-aligned for b128 copies

__global__ __launch_bounds__(256) void gate_gemm_kernel(
    const float* __restrict__ x_real, const float* __restrict__ x_imag,
    const __bf16* __restrict__ Wb, const float* __restrict__ b_gate,
    unsigned char* __restrict__ lvl_out) {
  __shared__ float  As[2][TM * AS_STRIDE];
  __shared__ __bf16 Bsh[2][8 * 512];     // 8 fragment tiles x 512 halves

  const int bm     = blockIdx.x;         // 0..127
  const int bn     = blockIdx.y;         // 0..7
  const int tid    = threadIdx.x;
  const int lane   = tid & 31;
  const int waveId = tid >> 5;
  const int wm     = waveId & 3;         // M quadrant (32 rows)
  const int wn     = waveId >> 2;        // N half (64 cols)

  auto copy_chunk = [&](int kc, float* Ad, __bf16* Bd) {
    const float* X  = (kc < DTOT) ? x_real : x_imag;
    const int kco   = (kc < DTOT) ? kc : kc - DTOT;
    // A tile: 128 x 32 f32 = 1024 float4 -> 4 async b128 per thread
    #pragma unroll
    for (int j = 0; j < 4; ++j) {
      int f = tid + j * 256;
      int r = f >> 3, c4 = (f & 7) << 2;
      g2l_16B(X + (size_t)(bm * TM + r) * DTOT + kco + c4,
              Ad + r * AS_STRIDE + c4);
    }
    // B: 8 contiguous fragment tiles (8KB) -> 2 async b128 per thread
    const __bf16* src = Wb + ((size_t)(kc >> 5) * 64 + bn * 8) * 512;
    #pragma unroll
    for (int j = 0; j < 2; ++j) {
      int f = tid + j * 256;             // 0..511
      g2l_16B(src + f * 8, Bd + f * 8);
    }
  };

  const v8f vz = {0.f, 0.f, 0.f, 0.f, 0.f, 0.f, 0.f, 0.f};
  v8f acc[2][4] = {{vz, vz, vz, vz}, {vz, vz, vz, vz}};

  copy_chunk(0, As[0], Bsh[0]);
  const int NCHUNK = KTOT / TK;          // 64
  for (int kk = 0; kk < NCHUNK; ++kk) {
    const int cur = kk & 1;
    if (kk + 1 < NCHUNK) {
      copy_chunk((kk + 1) * TK, As[cur ^ 1], Bsh[cur ^ 1]);
      WAIT_ASYNC(6);                     // oldest 6 (current buffer) done
    } else {
      WAIT_ASYNC(0);
    }
    __syncthreads();

    const float*  Ab = As[cur];
    const __bf16* Bb = Bsh[cur];

    // A frag (16x32 bf16): lane<16 -> row=lane, K 0..15;
    //                      lane>=16 -> row=lane-16, K 16..31
    // Batch all LDS reads first, then convert.
    float4 av[2][4];
    #pragma unroll
    for (int mt = 0; mt < 2; ++mt) {
      int row = wm * 32 + mt * 16 + (lane & 15);
      int kb  = (lane >> 4) * 16;
      const float4* p4 = (const float4*)(Ab + row * AS_STRIDE + kb);
      av[mt][0] = p4[0]; av[mt][1] = p4[1];   // K kb..kb+7
      av[mt][2] = p4[2]; av[mt][3] = p4[3];   // K kb+8..kb+15
    }
    v16bf a[2], b[4];
    #pragma unroll
    for (int mt = 0; mt < 2; ++mt) {
      #pragma unroll
      for (int q = 0; q < 4; ++q) {
        a[mt][q * 4 + 0] = (__bf16)av[mt][q].x;
        a[mt][q * 4 + 1] = (__bf16)av[mt][q].y;
        a[mt][q * 4 + 2] = (__bf16)av[mt][q].z;
        a[mt][q * 4 + 3] = (__bf16)av[mt][q].w;
      }
    }
    // B frags: one contiguous 32B load (2x b128) each from fragment-order tiles (wn*4+nt)
    #pragma unroll
    for (int nt = 0; nt < 4; ++nt)
      b[nt] = *(const v16bf*)(Bb + ((wn * 4 + nt) * 512 + lane * 16));

    #pragma unroll
    for (int mt = 0; mt < 2; ++mt)
      #pragma unroll
      for (int nt = 0; nt < 4; ++nt)
        acc[mt][nt] = __builtin_amdgcn_wmma_f32_16x16x32_bf16(
            false, a[mt], false, b[nt], (short)0, acc[mt][nt], false, false);
    __syncthreads();
  }

  // Epilogue: sigmoid, 8-column mean, quantize to level 0..7.
  // D layout: lane<16 -> col=lane, rows 0..7; lane>=16 -> col=lane-16, rows 8..15
  #pragma unroll
  for (int mt = 0; mt < 2; ++mt) {
    #pragma unroll
    for (int nt = 0; nt < 4; ++nt) {
      int colAbs  = bn * TN + wn * 64 + nt * 16 + (lane & 15);
      float bg    = b_gate[colAbs];
      int rowBase = bm * TM + wm * 32 + mt * 16 + (lane >> 4) * 8;
      #pragma unroll
      for (int r = 0; r < 8; ++r) {
        float z    = (acc[mt][nt][r] + bg) * 1.7015f;
        float gate = 1.0f / (1.0f + __expf(-z));
        gate += __shfl_xor(gate, 1, 32);
        gate += __shfl_xor(gate, 2, 32);
        gate += __shfl_xor(gate, 4, 32);
        if ((lane & 7) == 0) {
          float mean = gate * 0.125f;
          int lvl = (int)rintf(mean * 7.0f);
          lvl = lvl < 0 ? 0 : (lvl > 7 ? 7 : lvl);
          lvl_out[(size_t)(rowBase + r) * NBLK + (colAbs >> 3)] =
              (unsigned char)lvl;
        }
      }
    }
  }
}

// ---------------------------------------------------------------------------
// Kernel 3: apply the block-diagonal complex rotation + straight-through blend.
// Block = 4 n-blocks x 64 tokens; all 8 levels of U for those 4 blocks staged
// in LDS (16 KB). One thread per (token, block): 8x8 complex matvec.
// Output: complex64 interleaved (r,i).
// ---------------------------------------------------------------------------
__global__ __launch_bounds__(256) void apply_kernel(
    const float* __restrict__ x_real, const float* __restrict__ x_imag,
    const float* __restrict__ Ur, const float* __restrict__ Ui,
    const unsigned char* __restrict__ lvl_in, float* __restrict__ out) {
  __shared__ float Ulr[K_LEVELS * 4 * 64];
  __shared__ float Uli[K_LEVELS * 4 * 64];
  const int n0  = blockIdx.x * 4;
  const int m0  = blockIdx.y * 64;
  const int tid = threadIdx.x;

  #pragma unroll
  for (int j = 0; j < 8; ++j) {
    int e   = tid + j * 256;              // 0..2047
    int lvl = e >> 8;
    int nn  = (e >> 6) & 3;
    int q   = e & 63;
    int gi  = ((lvl << 7) + n0 + nn) * 64 + q;
    Ulr[e] = Ur[gi];
    Uli[e] = Ui[gi];
  }
  __syncthreads();

  const int nn = tid & 3;
  const int n  = n0 + nn;
  const int m  = m0 + (tid >> 2);
  const int lvl = lvl_in[(size_t)m * NBLK + n];
  const float gq = (float)lvl * (1.0f / 7.0f);
  const float s  = gq / (gq + 1e-7f);
  const float* ur = &Ulr[(lvl * 4 + nn) * 64];
  const float* ui = &Uli[(lvl * 4 + nn) * 64];

  float xr[8], xi[8];
  const float4* xr4 = (const float4*)(x_real + (size_t)m * DTOT + n * BSZ);
  const float4* xi4 = (const float4*)(x_imag + (size_t)m * DTOT + n * BSZ);
  ((float4*)xr)[0] = xr4[0];  ((float4*)xr)[1] = xr4[1];
  ((float4*)xi)[0] = xi4[0];  ((float4*)xi)[1] = xi4[1];

  float orv[8], oiv[8];
  #pragma unroll
  for (int o = 0; o < 8; ++o) {
    float ar = 0.f, ai = 0.f;
    #pragma unroll
    for (int i = 0; i < 8; ++i) {
      float u = ur[o * 8 + i], v = ui[o * 8 + i];
      ar += u * xr[i] - v * xi[i];
      ai += u * xi[i] + v * xr[i];
    }
    orv[o] = xr[o] + s * (ar - xr[o]);
    oiv[o] = xi[o] + s * (ai - xi[o]);
  }

  float4* op = (float4*)(out + 2 * ((size_t)m * DTOT + n * BSZ));
  #pragma unroll
  for (int o = 0; o < 4; ++o)
    op[o] = make_float4(orv[2 * o], oiv[2 * o], orv[2 * o + 1], oiv[2 * o + 1]);
}

// ---------------------------------------------------------------------------
extern "C" void kernel_launch(void* const* d_in, const int* in_sizes, int n_in,
                              void* d_out, int out_size, void* d_ws, size_t ws_size,
                              hipStream_t stream) {
  const float* x_real = (const float*)d_in[0];
  const float* x_imag = (const float*)d_in[1];
  const float* A_real = (const float*)d_in[2];
  const float* A_imag = (const float*)d_in[3];
  const float* W_gate = (const float*)d_in[4];
  const float* b_gate = (const float*)d_in[5];
  float* out = (float*)d_out;

  // Workspace: Ur (256KB) | Ui (256KB) | lvl (2MB) | Wb bf16 (4MB)
  float* Ur = (float*)d_ws;
  float* Ui = Ur + K_LEVELS * NBLK * 64;
  unsigned char* lvl = (unsigned char*)(Ui + K_LEVELS * NBLK * 64);
  __bf16* Wb = (__bf16*)(lvl + (size_t)MTOT * NBLK);

  wconv_kernel<<<(KTOT * DTOT) / 256, 256, 0, stream>>>(W_gate, Wb);
  cayley_kernel<<<K_LEVELS * NBLK, 32, 0, stream>>>(A_real, A_imag, Ur, Ui);
  gate_gemm_kernel<<<dim3(MTOT / TM, DTOT / TN), 256, 0, stream>>>(
      x_real, x_imag, Wb, b_gate, lvl);
  apply_kernel<<<dim3(NBLK / 4, MTOT / 64), 256, 0, stream>>>(
      x_real, x_imag, Ur, Ui, lvl, out);
}
